// single_task_2740189135403
// MI455X (gfx1250) — matrix operand order repaired and checked by reading, not compiled, not run on hardware
//
#include <hip/hip_runtime.h>

#define NN 50000
#define NE 800000
#define NG 256
#define FIN 64
#define HC 256
#define ODIM 128
#define ET (NE + NN)
#define LDA 80   // LDS row stride in halves (160B: multiple of 32B)

typedef __attribute__((ext_vector_type(16))) _Float16 v16h;
typedef __attribute__((ext_vector_type(8)))  _Float16 v8h;
typedef __attribute__((ext_vector_type(8)))  float    v8f;

__device__ __forceinline__ unsigned fkey(float f) {
    unsigned u = __float_as_uint(f);
    return (u & 0x80000000u) ? ~u : (u | 0x80000000u);
}
__device__ __forceinline__ float finv(unsigned k) {
    return (k & 0x80000000u) ? __uint_as_float(k & 0x7FFFFFFFu) : __uint_as_float(~k);
}

// ---------------- GEMM: xh[N,256] = x[N,64] @ lin_w[256,64]^T via WMMA f16 ----
__global__ __launch_bounds__(256) void k_gemm(const float* __restrict__ x,
                                              const float* __restrict__ lin_w,
                                              float* __restrict__ xh) {
    __shared__ __attribute__((aligned(32))) _Float16 As[64 * LDA];
    __shared__ __attribute__((aligned(32))) _Float16 Bs[256 * LDA];
    const int tid = threadIdx.x;
    const int m0 = blockIdx.x * 64;

    const int lrow = tid >> 2;
    const int lcol = (tid & 3) << 4;
    { // load A tile (64x64 f32 -> f16), coalesced float4
        int gm = m0 + lrow;
        if (gm < NN) {
            const float4* p = (const float4*)(x + (size_t)gm * FIN + lcol);
            #pragma unroll
            for (int j = 0; j < 4; ++j) {
                float4 v = p[j];
                As[lrow * LDA + lcol + j * 4 + 0] = (_Float16)v.x;
                As[lrow * LDA + lcol + j * 4 + 1] = (_Float16)v.y;
                As[lrow * LDA + lcol + j * 4 + 2] = (_Float16)v.z;
                As[lrow * LDA + lcol + j * 4 + 3] = (_Float16)v.w;
            }
        } else {
            #pragma unroll
            for (int j = 0; j < 16; ++j) As[lrow * LDA + lcol + j] = (_Float16)0.0f;
        }
    }
    // load B = lin_w (256x64), 4 passes of 64 rows, coalesced float4
    #pragma unroll
    for (int pss = 0; pss < 4; ++pss) {
        int row = pss * 64 + lrow;
        const float4* q = (const float4*)(lin_w + (size_t)row * FIN + lcol);
        #pragma unroll
        for (int j = 0; j < 4; ++j) {
            float4 v = q[j];
            Bs[row * LDA + lcol + j * 4 + 0] = (_Float16)v.x;
            Bs[row * LDA + lcol + j * 4 + 1] = (_Float16)v.y;
            Bs[row * LDA + lcol + j * 4 + 2] = (_Float16)v.z;
            Bs[row * LDA + lcol + j * 4 + 3] = (_Float16)v.w;
        }
    }
    __syncthreads();

    const int wave = tid >> 5, lane = tid & 31;
    const int laneLo = lane & 15, laneHi = lane >> 4;
    for (int t = 0; t < 8; ++t) {
        int tile = t * 8 + wave;          // 64 tiles: 4 M-tiles x 16 N-tiles
        int mt = tile >> 4, nt = tile & 15;
        int arow = mt * 16 + laneLo;
        int bn = nt * 16 + laneLo;
        v8f c = {};
        #pragma unroll
        for (int kb = 0; kb < 64; kb += 32) {
            // A frag: row=laneLo, halves 0..7 -> K = kb+laneHi*8.., halves 8..15 -> K = kb+16+laneHi*8..
            v8h alo = *(const v8h*)&As[arow * LDA + kb + laneHi * 8];
            v8h ahi = *(const v8h*)&As[arow * LDA + kb + 16 + laneHi * 8];
            v16h a;
            #pragma unroll
            for (int i = 0; i < 8; ++i) { a[i] = alo[i]; a[i + 8] = ahi[i]; }
            // B frag: N=laneLo, halves 0..7 -> K = kb+laneHi*8.., halves 8..15 -> K = kb+16+laneHi*8..
            v8h blo = *(const v8h*)&Bs[bn * LDA + kb + laneHi * 8];
            v8h bhi = *(const v8h*)&Bs[bn * LDA + kb + 16 + laneHi * 8];
            v16h b;
            #pragma unroll
            for (int i = 0; i < 8; ++i) { b[i] = blo[i]; b[i + 8] = bhi[i]; }
            c = __builtin_amdgcn_wmma_f32_16x16x32_f16(false, a, false, b, (short)0, c, false, false);
        }
        // C: M = r + 8*laneHi, N = laneLo.
        // NN is a multiple of 16 and tiles start at multiples of 16, so a tile
        // is either fully in range or fully out -> one scalar branch, 8
        // unguarded stores from a single base pointer with immediate offsets.
        int tstart = m0 + mt * 16;
        if (__builtin_amdgcn_readfirstlane(tstart) < NN) {
            float* outp = xh + (size_t)(tstart + 8 * laneHi) * HC + nt * 16 + laneLo;
            #pragma unroll
            for (int r = 0; r < 8; ++r) outp[(size_t)r * HC] = c[r];
        }
    }
}

// ------------- per-node attention logits a_src/a_dst [N,4] ------------------
__global__ __launch_bounds__(256) void k_attdot(const float* __restrict__ xh,
                                                const float* __restrict__ att_src,
                                                const float* __restrict__ att_dst,
                                                float* __restrict__ a_src,
                                                float* __restrict__ a_dst) {
    int g = blockIdx.x * 256 + threadIdx.x;
    if (g >= NN * 4) return;
    int node = g >> 2, h = g & 3;
    const float4* p = (const float4*)(xh + (size_t)node * HC + h * FIN);
    const float4* ps = (const float4*)(att_src + h * FIN);
    const float4* pd = (const float4*)(att_dst + h * FIN);
    float s = 0.f, d = 0.f;
    #pragma unroll
    for (int j = 0; j < 16; ++j) {
        float4 v = p[j], a = ps[j], b = pd[j];
        s += v.x * a.x + v.y * a.y + v.z * a.z + v.w * a.w;
        d += v.x * b.x + v.y * b.y + v.z * b.z + v.w * b.w;
    }
    a_src[g] = s; a_dst[g] = d;
}

// ------------- edge pass 1: raw leaky-relu alpha + segment max --------------
__global__ __launch_bounds__(256) void k_alpha_max(const int* __restrict__ ei,
                                                   const float* __restrict__ a_src,
                                                   const float* __restrict__ a_dst,
                                                   float4* __restrict__ alpha4,
                                                   unsigned* __restrict__ amaxk) {
    int e = blockIdx.x * 256 + threadIdx.x;
    if (e >= ET) return;
    int src = (e < NE) ? ei[e] : (e - NE);
    int dst = (e < NE) ? ei[NE + e] : (e - NE);
    const float4 s4 = *(const float4*)(a_src + src * 4);
    const float4 d4 = *(const float4*)(a_dst + dst * 4);
    float4 a;
    a.x = s4.x + d4.x; a.x = (a.x > 0.f) ? a.x : 0.2f * a.x;
    a.y = s4.y + d4.y; a.y = (a.y > 0.f) ? a.y : 0.2f * a.y;
    a.z = s4.z + d4.z; a.z = (a.z > 0.f) ? a.z : 0.2f * a.z;
    a.w = s4.w + d4.w; a.w = (a.w > 0.f) ? a.w : 0.2f * a.w;
    alpha4[e] = a;
    atomicMax(&amaxk[dst * 4 + 0], fkey(a.x));
    atomicMax(&amaxk[dst * 4 + 1], fkey(a.y));
    atomicMax(&amaxk[dst * 4 + 2], fkey(a.z));
    atomicMax(&amaxk[dst * 4 + 3], fkey(a.w));
}

// ------------- edge pass 2: exp(alpha - max) + segment sum ------------------
__global__ __launch_bounds__(256) void k_exp_sum(const int* __restrict__ ei,
                                                 float4* __restrict__ alpha4,
                                                 const unsigned* __restrict__ amaxk,
                                                 float* __restrict__ denom) {
    int e = blockIdx.x * 256 + threadIdx.x;
    if (e >= ET) return;
    int dst = (e < NE) ? ei[NE + e] : (e - NE);
    float4 a = alpha4[e];
    a.x = __expf(a.x - finv(amaxk[dst * 4 + 0]));
    a.y = __expf(a.y - finv(amaxk[dst * 4 + 1]));
    a.z = __expf(a.z - finv(amaxk[dst * 4 + 2]));
    a.w = __expf(a.w - finv(amaxk[dst * 4 + 3]));
    alpha4[e] = a;
    atomicAdd(&denom[dst * 4 + 0], a.x);
    atomicAdd(&denom[dst * 4 + 1], a.y);
    atomicAdd(&denom[dst * 4 + 2], a.z);
    atomicAdd(&denom[dst * 4 + 3], a.w);
}

// ------------- edge pass 3: normalize alpha in place ------------------------
__global__ __launch_bounds__(256) void k_norm_alpha(const int* __restrict__ ei,
                                                    float4* __restrict__ alpha4,
                                                    const float* __restrict__ denom) {
    int e = blockIdx.x * 256 + threadIdx.x;
    if (e >= ET) return;
    int dst = (e < NE) ? ei[NE + e] : (e - NE);
    float4 a = alpha4[e];
    a.x /= (denom[dst * 4 + 0] + 1e-16f);
    a.y /= (denom[dst * 4 + 1] + 1e-16f);
    a.z /= (denom[dst * 4 + 2] + 1e-16f);
    a.w /= (denom[dst * 4 + 3] + 1e-16f);
    alpha4[e] = a;
}

// ------------- edge pass 4: scatter messages (wave per edge) ----------------
__global__ __launch_bounds__(256) void k_scatter(const int* __restrict__ ei,
                                                 const float* __restrict__ xh,
                                                 const float* __restrict__ alphaOut,
                                                 float* __restrict__ accum) {
    int e = (blockIdx.x * 256 + threadIdx.x) >> 5;
    int lane = threadIdx.x & 31;
    if (e >= ET) return;
    int src = (e < NE) ? ei[e] : (e - NE);
    int dst = (e < NE) ? ei[NE + e] : (e - NE);
    int head = lane >> 3;                       // lane owns channels lane*8..+7 (one head)
    float a = alphaOut[(size_t)e * 4 + head];
    int c0 = lane * 8;
    const float4* p = (const float4*)(xh + (size_t)src * HC + c0);
    float4 v0 = p[0], v1 = p[1];
    float* ab = accum + (size_t)dst * HC + c0;
    atomicAdd(ab + 0, v0.x * a); atomicAdd(ab + 1, v0.y * a);
    atomicAdd(ab + 2, v0.z * a); atomicAdd(ab + 3, v0.w * a);
    atomicAdd(ab + 4, v1.x * a); atomicAdd(ab + 5, v1.y * a);
    atomicAdd(ab + 6, v1.z * a); atomicAdd(ab + 7, v1.w * a);
}

// ------------- GraphNorm stage 1: x2 = accum + bias; graph sums -------------
__global__ __launch_bounds__(256) void k_gn1(const int* __restrict__ batch,
                                             const float* __restrict__ gat_bias,
                                             float* __restrict__ w1,
                                             float* __restrict__ gsum,
                                             float* __restrict__ gcnt) {
    int node = blockIdx.x, c = threadIdx.x;
    int g = batch[node];
    size_t idx = (size_t)node * HC + c;
    float v = w1[idx] + gat_bias[c];
    w1[idx] = v;
    atomicAdd(&gsum[g * HC + c], v);
    if (c == 0) atomicAdd(&gcnt[g], 1.0f);
}

__global__ __launch_bounds__(256) void k_mean(const float* __restrict__ gsum,
                                              const float* __restrict__ gcnt,
                                              float* __restrict__ mean) {
    int g = blockIdx.x, c = threadIdx.x;
    float cnt = gcnt[g]; if (cnt < 1.f) cnt = 1.f;
    mean[g * HC + c] = gsum[g * HC + c] / cnt;
}

__global__ __launch_bounds__(256) void k_gn2(const int* __restrict__ batch,
                                             const float* __restrict__ mean,
                                             const float* __restrict__ mscale,
                                             float* __restrict__ w1,
                                             float* __restrict__ gvar) {
    int node = blockIdx.x, c = threadIdx.x;
    int g = batch[node];
    size_t idx = (size_t)node * HC + c;
    float sub = w1[idx] - mean[g * HC + c] * mscale[c];
    w1[idx] = sub;
    atomicAdd(&gvar[g * HC + c], sub * sub);
}

__global__ __launch_bounds__(256) void k_rstd(const float* __restrict__ gvar,
                                              const float* __restrict__ gcnt,
                                              float* __restrict__ rstd) {
    int g = blockIdx.x, c = threadIdx.x;
    float cnt = gcnt[g]; if (cnt < 1.f) cnt = 1.f;
    rstd[g * HC + c] = rsqrtf(gvar[g * HC + c] / cnt + 1e-5f);
}

__global__ __launch_bounds__(256) void k_gn3(const int* __restrict__ batch,
                                             const float* __restrict__ gw,
                                             const float* __restrict__ gb,
                                             const float* __restrict__ rstd,
                                             float* __restrict__ w1) {
    int node = blockIdx.x, c = threadIdx.x;
    int g = batch[node];
    size_t idx = (size_t)node * HC + c;
    float h = gw[c] * w1[idx] * rstd[g * HC + c] + gb[c];
    w1[idx] = fmaxf(h, 0.f);                    // ReLU
}

// ------------- gate_nn: sigmoid(relu(h@W1^T+b1)@W2^T+b2); graph max ---------
__global__ __launch_bounds__(256) void k_gate(const int* __restrict__ batch,
                                              const float* __restrict__ w1,
                                              const float* __restrict__ att1_w,
                                              const float* __restrict__ att1_b,
                                              const float* __restrict__ att2_w,
                                              const float* __restrict__ att2_b,
                                              float* __restrict__ gate,
                                              unsigned* __restrict__ gmaxk) {
    __shared__ float w1s[16 * HC];
    for (int i = threadIdx.x; i < 16 * HC; i += 256) w1s[i] = att1_w[i];
    __syncthreads();
    int node = blockIdx.x * 256 + threadIdx.x;
    if (node >= NN) return;
    float acc[16];
    #pragma unroll
    for (int j = 0; j < 16; ++j) acc[j] = att1_b[j];
    const float4* p = (const float4*)(w1 + (size_t)node * HC);
    for (int c4 = 0; c4 < 64; ++c4) {
        float4 v = p[c4];
        int c = c4 * 4;
        #pragma unroll
        for (int j = 0; j < 16; ++j)
            acc[j] += v.x * w1s[j * HC + c] + v.y * w1s[j * HC + c + 1] +
                      v.z * w1s[j * HC + c + 2] + v.w * w1s[j * HC + c + 3];
    }
    float gp = att2_b[0];
    #pragma unroll
    for (int j = 0; j < 16; ++j) gp += fmaxf(acc[j], 0.f) * att2_w[j];
    float gt = 1.f / (1.f + __expf(-gp));
    gate[node] = gt;
    atomicMax(&gmaxk[batch[node]], fkey(gt));
}

__global__ __launch_bounds__(256) void k_gatenorm(const int* __restrict__ batch,
                                                  float* __restrict__ gate,
                                                  const unsigned* __restrict__ gmaxk,
                                                  float* __restrict__ gden) {
    int node = blockIdx.x * 256 + threadIdx.x;
    if (node >= NN) return;
    int g = batch[node];
    float ex = __expf(gate[node] - finv(gmaxk[g]));
    gate[node] = ex;
    atomicAdd(&gden[g], ex);
}

__global__ __launch_bounds__(256) void k_pool(const int* __restrict__ batch,
                                              const float* __restrict__ gate,
                                              const float* __restrict__ gden,
                                              const float* __restrict__ w1,
                                              float* __restrict__ pooled) {
    int node = blockIdx.x, c = threadIdx.x;
    int g = batch[node];
    float w = gate[node] / (gden[g] + 1e-16f);
    atomicAdd(&pooled[g * HC + c], w * w1[(size_t)node * HC + c]);
}

// ------------- final heads: sigmoid(relu(pooled@fc1^T+b)@out_w^T+b) ---------
__global__ __launch_bounds__(128) void k_head(const float* __restrict__ pooled,
                                              const float* __restrict__ fc1_w,
                                              const float* __restrict__ fc1_b,
                                              const float* __restrict__ out_w,
                                              const float* __restrict__ out_b,
                                              float* __restrict__ out) {
    __shared__ float ps[HC];
    __shared__ float red[ODIM];
    int g = blockIdx.x, j = threadIdx.x;
    ps[j] = pooled[g * HC + j];
    ps[j + 128] = pooled[g * HC + j + 128];
    __syncthreads();
    float acc = fc1_b[j];
    for (int c = 0; c < HC; ++c) acc += ps[c] * fc1_w[j * HC + c];
    red[j] = fmaxf(acc, 0.f) * out_w[j];
    __syncthreads();
    for (int s = 64; s > 0; s >>= 1) {
        if (j < s) red[j] += red[j + s];
        __syncthreads();
    }
    if (j == 0) out[g] = 1.f / (1.f + __expf(-(red[0] + out_b[0])));
}

extern "C" void kernel_launch(void* const* d_in, const int* in_sizes, int n_in,
                              void* d_out, int out_size, void* d_ws, size_t ws_size,
                              hipStream_t stream) {
    const float* x        = (const float*)d_in[0];
    const int*   ei       = (const int*)d_in[1];
    const int*   batch    = (const int*)d_in[2];
    const float* lin_w    = (const float*)d_in[3];
    const float* att_src  = (const float*)d_in[4];
    const float* att_dst  = (const float*)d_in[5];
    const float* gat_bias = (const float*)d_in[6];
    const float* gn_w     = (const float*)d_in[7];
    const float* gn_b     = (const float*)d_in[8];
    const float* gn_ms    = (const float*)d_in[9];
    const float* fc1_w    = (const float*)d_in[10];
    const float* fc1_b    = (const float*)d_in[11];
    const float* out_w    = (const float*)d_in[12];
    const float* out_b    = (const float*)d_in[13];
    const float* att1_w   = (const float*)d_in[14];
    const float* att1_b   = (const float*)d_in[15];
    const float* att2_w   = (const float*)d_in[16];
    const float* att2_b   = (const float*)d_in[17];

    float* out = (float*)d_out;
    float* alphaOut = out + NG;                 // alpha [ET,4] follows out [256]
    float4* alpha4 = (float4*)alphaOut;         // byte offset 1024: 16B aligned

    char* w = (char*)d_ws;
    size_t off = 0;
    auto alloc = [&](size_t bytes) -> void* {
        void* p = w + off;
        off += (bytes + 255) & ~(size_t)255;
        return p;
    };
    float*    xh     = (float*)alloc((size_t)NN * HC * 4);
    float*    w1     = (float*)alloc((size_t)NN * HC * 4);   // accum -> x2 -> sub -> h
    float*    a_src  = (float*)alloc((size_t)NN * 4 * 4);
    float*    a_dst  = (float*)alloc((size_t)NN * 4 * 4);
    unsigned* amaxk  = (unsigned*)alloc((size_t)NN * 4 * 4);
    float*    denom  = (float*)alloc((size_t)NN * 4 * 4);
    float*    gsum   = (float*)alloc((size_t)NG * HC * 4);
    float*    gcnt   = (float*)alloc((size_t)NG * 4);
    float*    mean   = (float*)alloc((size_t)NG * HC * 4);
    float*    gvar   = (float*)alloc((size_t)NG * HC * 4);
    float*    rstd   = (float*)alloc((size_t)NG * HC * 4);
    float*    gate   = (float*)alloc((size_t)NN * 4);
    unsigned* gmaxk  = (unsigned*)alloc((size_t)NG * 4);
    float*    gden   = (float*)alloc((size_t)NG * 4);
    float*    pooled = (float*)alloc((size_t)NG * HC * 4);

    hipMemsetAsync(w1, 0, (size_t)NN * HC * 4, stream);
    hipMemsetAsync(amaxk, 0, (size_t)NN * 4 * 4, stream);
    hipMemsetAsync(denom, 0, (size_t)NN * 4 * 4, stream);
    hipMemsetAsync(gsum, 0, (size_t)NG * HC * 4, stream);
    hipMemsetAsync(gcnt, 0, (size_t)NG * 4, stream);
    hipMemsetAsync(gvar, 0, (size_t)NG * HC * 4, stream);
    hipMemsetAsync(gmaxk, 0, (size_t)NG * 4, stream);
    hipMemsetAsync(gden, 0, (size_t)NG * 4, stream);
    hipMemsetAsync(pooled, 0, (size_t)NG * HC * 4, stream);

    k_gemm<<<(NN + 63) / 64, 256, 0, stream>>>(x, lin_w, xh);
    k_attdot<<<(NN * 4 + 255) / 256, 256, 0, stream>>>(xh, att_src, att_dst, a_src, a_dst);
    k_alpha_max<<<(ET + 255) / 256, 256, 0, stream>>>(ei, a_src, a_dst, alpha4, amaxk);
    k_exp_sum<<<(ET + 255) / 256, 256, 0, stream>>>(ei, alpha4, amaxk, denom);
    k_norm_alpha<<<(ET + 255) / 256, 256, 0, stream>>>(ei, alpha4, denom);
    k_scatter<<<(ET + 7) / 8, 256, 0, stream>>>(ei, xh, alphaOut, w1);
    k_gn1<<<NN, 256, 0, stream>>>(batch, gat_bias, w1, gsum, gcnt);
    k_mean<<<NG, 256, 0, stream>>>(gsum, gcnt, mean);
    k_gn2<<<NN, 256, 0, stream>>>(batch, mean, gn_ms, w1, gvar);
    k_rstd<<<NG, 256, 0, stream>>>(gvar, gcnt, rstd);
    k_gn3<<<NN, 256, 0, stream>>>(batch, gn_w, gn_b, rstd, w1);
    k_gate<<<(NN + 255) / 256, 256, 0, stream>>>(batch, w1, att1_w, att1_b, att2_w, att2_b, gate, gmaxk);
    k_gatenorm<<<(NN + 255) / 256, 256, 0, stream>>>(batch, gate, gmaxk, gden);
    k_pool<<<NN, 256, 0, stream>>>(batch, gate, gden, w1, pooled);
    k_head<<<NG, 128, 0, stream>>>(pooled, fc1_w, fc1_b, out_w, out_b, out);
}
